// SkipAttention_13005160972362
// MI455X (gfx1250) — matrix operand order repaired and checked by reading, not compiled, not run on hardware
//
#include <hip/hip_runtime.h>

// ---------------------------------------------------------------------------
// Types for CDNA5 WMMA (wave32): v_wmma_f32_16x16x32_bf16
// ---------------------------------------------------------------------------
typedef __bf16 bf16_t;
typedef bf16_t bf16x8_t  __attribute__((ext_vector_type(8)));
typedef bf16_t bf16x16_t __attribute__((ext_vector_type(16)));
typedef float  f32x8_t   __attribute__((ext_vector_type(8)));

constexpr int kB = 8;      // batch
constexpr int kN = 2048;   // queries
constexpr int kM = 2048;   // keys
constexpr int kD = 512;    // feature dim
constexpr int kRows = kB * kN;          // 16384 flat rows (== kB*kM)
constexpr int kPStride = 2 * kM;        // bf16 attn row stride (aliased over f32 S)

// native cast -> hardware v_cvt_*_bf16_f32
__device__ __forceinline__ bf16_t f2bf(float f) { return (bf16_t)f; }

__device__ __forceinline__ f32x8_t zero8() {
  f32x8_t z = {0.f, 0.f, 0.f, 0.f, 0.f, 0.f, 0.f, 0.f};
  return z;
}

__device__ __forceinline__ f32x8_t wmma_bf16(bf16x16_t a, bf16x16_t b, f32x8_t c) {
  return __builtin_amdgcn_wmma_f32_16x16x32_bf16(false, a, false, b, (short)0, c,
                                                 false, false);
}

// A fragment (16x32, M x K): lane 0-15 -> row M=lane, K = {k0..k0+7} u {k0+16..k0+23}
//                            lane 16-31 -> K shifted by +8.
__device__ __forceinline__ bf16x16_t frag_a_bf16(const bf16_t* rowp, int k0, int half) {
  const bf16_t* p0 = rowp + k0 + half * 8;
  bf16x8_t lo = *(const bf16x8_t*)p0;
  bf16x8_t hi = *(const bf16x8_t*)(p0 + 16);
  bf16x16_t r;
#pragma unroll
  for (int i = 0; i < 8; ++i) { r[i] = lo[i]; r[8 + i] = hi[i]; }
  return r;
}

// B fragment (32x16, K x N), src stored row-major with the N index as the row:
// lane 0-15 -> col N=lane, K = {k0..k0+7} u {k0+16..k0+23}; lane 16-31 -> K shifted by +8.
__device__ __forceinline__ bf16x16_t frag_b_bf16(const bf16_t* colp, int k0, int half) {
  const bf16_t* p = colp + k0 + half * 8;
  bf16x8_t lo = *(const bf16x8_t*)p;
  bf16x8_t hi = *(const bf16x8_t*)(p + 16);
  bf16x16_t r;
#pragma unroll
  for (int i = 0; i < 8; ++i) { r[i] = lo[i]; r[8 + i] = hi[i]; }
  return r;
}

// ---------------------------------------------------------------------------
// K0: bulk f32 -> bf16 conversion (feats + weights), 8 elements/thread
// ---------------------------------------------------------------------------
__global__ void __launch_bounds__(256)
cvt_kernel(const float* __restrict__ in, bf16_t* __restrict__ out, int n8) {
  const int i = blockIdx.x * 256 + threadIdx.x;
  if (i >= n8) return;
  const float4 a = ((const float4*)in)[2 * i];
  const float4 b = ((const float4*)in)[2 * i + 1];
  bf16x8_t r;
  r[0] = f2bf(a.x); r[1] = f2bf(a.y); r[2] = f2bf(a.z); r[3] = f2bf(a.w);
  r[4] = f2bf(b.x); r[5] = f2bf(b.y); r[6] = f2bf(b.z); r[7] = f2bf(b.w);
  ((bf16x8_t*)out)[i] = r;
}

// ---------------------------------------------------------------------------
// K1: projection  out = act @ W^T + bias  -> bf16 (optionally transposed per batch)
// block = 256 thr (8 waves), block tile 128 rows x 128 cols, wave tile 32x64
// ---------------------------------------------------------------------------
template <bool TRANS>
__global__ void __launch_bounds__(256)
proj_kernel(const bf16_t* __restrict__ act, const bf16_t* __restrict__ W,
            const float* __restrict__ bias, bf16_t* __restrict__ out,
            int rowsPerBatch) {
  const int lane = threadIdx.x & 31;
  const int wave = threadIdx.x >> 5;
  const int wr = wave >> 1;            // 0..3
  const int wc = wave & 1;             // 0..1
  const int half = lane >> 4;
  const int lidx = lane & 15;

  const int rowBlk = blockIdx.x * 128 + wr * 32;
  const int colBlk = blockIdx.y * 128 + wc * 64;

  f32x8_t acc[2][4];
#pragma unroll
  for (int i = 0; i < 2; ++i)
#pragma unroll
    for (int j = 0; j < 4; ++j) acc[i][j] = zero8();

  const bf16_t* arow[2];
  const bf16_t* bcol[4];
#pragma unroll
  for (int tr = 0; tr < 2; ++tr)
    arow[tr] = act + (size_t)(rowBlk + tr * 16 + lidx) * kD;
#pragma unroll
  for (int tc = 0; tc < 4; ++tc)
    bcol[tc] = W + (size_t)(colBlk + tc * 16 + lidx) * kD;

  for (int k0 = 0; k0 < kD; k0 += 32) {
    bf16x16_t a[2], b[4];
#pragma unroll
    for (int tr = 0; tr < 2; ++tr) a[tr] = frag_a_bf16(arow[tr], k0, half);
#pragma unroll
    for (int tc = 0; tc < 4; ++tc) b[tc] = frag_b_bf16(bcol[tc], k0, half);
#pragma unroll
    for (int tr = 0; tr < 2; ++tr)
#pragma unroll
      for (int tc = 0; tc < 4; ++tc) acc[tr][tc] = wmma_bf16(a[tr], b[tc], acc[tr][tc]);
  }

#pragma unroll
  for (int tr = 0; tr < 2; ++tr)
#pragma unroll
    for (int tc = 0; tc < 4; ++tc) {
      const int col = colBlk + tc * 16 + lidx;
      const float bv = bias[col];
#pragma unroll
      for (int r = 0; r < 8; ++r) {
        const int row = rowBlk + tr * 16 + r + 8 * half;
        const float v = acc[tr][tc][r] + bv;
        if (TRANS) {
          const int bb = row / rowsPerBatch;
          const int ml = row - bb * rowsPerBatch;
          out[((size_t)bb * kD + col) * rowsPerBatch + ml] = f2bf(v);
        } else {
          out[(size_t)row * kD + col] = f2bf(v);
        }
      }
    }
}

// ---------------------------------------------------------------------------
// K2: scores S = (Q K^T)/sqrt(D) + log(exp(-2*dist)+1e-8)   -> f32
// block tile 64 n x 256 m, wave tile 32x64; xyz staged in LDS
// ---------------------------------------------------------------------------
__global__ void __launch_bounds__(256)
scores_kernel(const bf16_t* __restrict__ Qb, const bf16_t* __restrict__ Kb,
              const float* __restrict__ qxyz, const float* __restrict__ kxyz,
              float* __restrict__ S) {
  __shared__ float sQ[64 * 3];
  __shared__ float sK[256 * 3];

  const int bb = blockIdx.z;
  const int n0 = blockIdx.y * 64;
  const int m0 = blockIdx.x * 256;
  const int t = threadIdx.x;

  if (t < 64) {
    const float* p = qxyz + ((size_t)bb * kN + n0 + t) * 3;
    sQ[t * 3 + 0] = p[0];
    sQ[t * 3 + 1] = p[1];
    sQ[t * 3 + 2] = p[2];
  }
  {
    const float* p = kxyz + ((size_t)bb * kM + m0 + t) * 3;
    sK[t * 3 + 0] = p[0];
    sK[t * 3 + 1] = p[1];
    sK[t * 3 + 2] = p[2];
  }
  __syncthreads();

  const int lane = t & 31;
  const int wave = t >> 5;
  const int wr = wave >> 2;            // 0..1
  const int wc = wave & 3;             // 0..3
  const int half = lane >> 4;
  const int lidx = lane & 15;

  f32x8_t acc[2][4];
#pragma unroll
  for (int i = 0; i < 2; ++i)
#pragma unroll
    for (int j = 0; j < 4; ++j) acc[i][j] = zero8();

  const bf16_t* arow[2];
  const bf16_t* bcol[4];
#pragma unroll
  for (int tr = 0; tr < 2; ++tr)
    arow[tr] = Qb + ((size_t)bb * kN + n0 + wr * 32 + tr * 16 + lidx) * kD;
#pragma unroll
  for (int tc = 0; tc < 4; ++tc)
    bcol[tc] = Kb + ((size_t)bb * kM + m0 + wc * 64 + tc * 16 + lidx) * kD;

  for (int k0 = 0; k0 < kD; k0 += 32) {
    bf16x16_t a[2], b[4];
#pragma unroll
    for (int tr = 0; tr < 2; ++tr) a[tr] = frag_a_bf16(arow[tr], k0, half);
#pragma unroll
    for (int tc = 0; tc < 4; ++tc) b[tc] = frag_b_bf16(bcol[tc], k0, half);
#pragma unroll
    for (int tr = 0; tr < 2; ++tr)
#pragma unroll
      for (int tc = 0; tc < 4; ++tc) acc[tr][tc] = wmma_bf16(a[tr], b[tc], acc[tr][tc]);
  }

  const float scale = 0.044194173824159216f;  // 1/sqrt(512)
#pragma unroll
  for (int tr = 0; tr < 2; ++tr)
#pragma unroll
    for (int tc = 0; tc < 4; ++tc) {
      const int ml = wc * 64 + tc * 16 + lidx;
      const float kx = sK[ml * 3 + 0], ky = sK[ml * 3 + 1], kz = sK[ml * 3 + 2];
#pragma unroll
      for (int r = 0; r < 8; ++r) {
        const int nl = wr * 32 + tr * 16 + r + 8 * half;
        const float dx = sQ[nl * 3 + 0] - kx;
        const float dy = sQ[nl * 3 + 1] - ky;
        const float dz = sQ[nl * 3 + 2] - kz;
        const float d2 = fmaxf(dx * dx + dy * dy + dz * dz, 1e-12f);
        const float dist = sqrtf(d2);
        const float bias = __logf(__expf(-2.0f * dist) + 1e-8f);
        S[((size_t)bb * kN + n0 + nl) * kM + (m0 + ml)] = acc[tr][tc][r] * scale + bias;
      }
    }
}

// ---------------------------------------------------------------------------
// K3: row softmax; writes bf16 probabilities in place over the f32 row
// (bf16 row stride = 2*kM elements)
// ---------------------------------------------------------------------------
__global__ void __launch_bounds__(256)
softmax_kernel(float* __restrict__ S) {
  __shared__ float red[256];
  const size_t row = blockIdx.x;
  float* p = S + row * (size_t)kM;
  bf16_t* o = (bf16_t*)p;
  const int t = threadIdx.x;

  float x[8];
#pragma unroll
  for (int i = 0; i < 8; ++i) x[i] = p[t + i * 256];

  float m = -3.4e38f;
#pragma unroll
  for (int i = 0; i < 8; ++i) m = fmaxf(m, x[i]);
  red[t] = m;
  __syncthreads();
  for (int s = 128; s > 0; s >>= 1) {
    if (t < s) red[t] = fmaxf(red[t], red[t + s]);
    __syncthreads();
  }
  const float rowmax = red[0];
  __syncthreads();

  float e[8];
  float sum = 0.f;
#pragma unroll
  for (int i = 0; i < 8; ++i) {
    e[i] = __expf(x[i] - rowmax);
    sum += e[i];
  }
  red[t] = sum;
  __syncthreads();
  for (int s = 128; s > 0; s >>= 1) {
    if (t < s) red[t] += red[t + s];
    __syncthreads();
  }
  const float inv = 1.0f / red[0];
  __syncthreads();  // all reads of the f32 row are complete before bf16 writes

#pragma unroll
  for (int i = 0; i < 8; ++i) o[t + i * 256] = f2bf(e[i] * inv);
}

// ---------------------------------------------------------------------------
// K4: attended = P @ V  (P bf16 aliased over S, stride 2*kM; V stored transposed)
// block tile 128 n x 128 e, wave tile 32x64, K loop over kM
// ---------------------------------------------------------------------------
__global__ void __launch_bounds__(256)
pv_kernel(const bf16_t* __restrict__ P, const bf16_t* __restrict__ Vt,
          bf16_t* __restrict__ att) {
  const int lane = threadIdx.x & 31;
  const int wave = threadIdx.x >> 5;
  const int wr = wave >> 1;
  const int wc = wave & 1;
  const int half = lane >> 4;
  const int lidx = lane & 15;

  const int rowBlk = blockIdx.x * 128 + wr * 32;   // global (b*N+n)
  const int colBlk = blockIdx.y * 128 + wc * 64;   // e
  const int bb = (blockIdx.x * 128) / kN;

  f32x8_t acc[2][4];
#pragma unroll
  for (int i = 0; i < 2; ++i)
#pragma unroll
    for (int j = 0; j < 4; ++j) acc[i][j] = zero8();

  const bf16_t* arow[2];
  const bf16_t* bcol[4];
#pragma unroll
  for (int tr = 0; tr < 2; ++tr)
    arow[tr] = P + (size_t)(rowBlk + tr * 16 + lidx) * kPStride;
#pragma unroll
  for (int tc = 0; tc < 4; ++tc)
    bcol[tc] = Vt + ((size_t)bb * kD + colBlk + tc * 16 + lidx) * kM;

  for (int k0 = 0; k0 < kM; k0 += 32) {
    bf16x16_t a[2], b[4];
#pragma unroll
    for (int tr = 0; tr < 2; ++tr) a[tr] = frag_a_bf16(arow[tr], k0, half);
#pragma unroll
    for (int tc = 0; tc < 4; ++tc) b[tc] = frag_b_bf16(bcol[tc], k0, half);
#pragma unroll
    for (int tr = 0; tr < 2; ++tr)
#pragma unroll
      for (int tc = 0; tc < 4; ++tc) acc[tr][tc] = wmma_bf16(a[tr], b[tc], acc[tr][tc]);
  }

#pragma unroll
  for (int tr = 0; tr < 2; ++tr)
#pragma unroll
    for (int tc = 0; tc < 4; ++tc) {
      const int col = colBlk + tc * 16 + lidx;
#pragma unroll
      for (int r = 0; r < 8; ++r) {
        const int row = rowBlk + tr * 16 + r + 8 * half;
        att[(size_t)row * kD + col] = f2bf(acc[tr][tc][r]);
      }
    }
}

// ---------------------------------------------------------------------------
// K5: y = resid + att @ Wo^T + bo   -> f32
// ---------------------------------------------------------------------------
__global__ void __launch_bounds__(256)
outproj_kernel(const bf16_t* __restrict__ act, const bf16_t* __restrict__ W,
               const float* __restrict__ bias, const float* __restrict__ resid,
               float* __restrict__ y) {
  const int lane = threadIdx.x & 31;
  const int wave = threadIdx.x >> 5;
  const int wr = wave >> 1;
  const int wc = wave & 1;
  const int half = lane >> 4;
  const int lidx = lane & 15;

  const int rowBlk = blockIdx.x * 128 + wr * 32;
  const int colBlk = blockIdx.y * 128 + wc * 64;

  f32x8_t acc[2][4];
#pragma unroll
  for (int i = 0; i < 2; ++i)
#pragma unroll
    for (int j = 0; j < 4; ++j) acc[i][j] = zero8();

  const bf16_t* arow[2];
  const bf16_t* bcol[4];
#pragma unroll
  for (int tr = 0; tr < 2; ++tr)
    arow[tr] = act + (size_t)(rowBlk + tr * 16 + lidx) * kD;
#pragma unroll
  for (int tc = 0; tc < 4; ++tc)
    bcol[tc] = W + (size_t)(colBlk + tc * 16 + lidx) * kD;

  for (int k0 = 0; k0 < kD; k0 += 32) {
    bf16x16_t a[2], b[4];
#pragma unroll
    for (int tr = 0; tr < 2; ++tr) a[tr] = frag_a_bf16(arow[tr], k0, half);
#pragma unroll
    for (int tc = 0; tc < 4; ++tc) b[tc] = frag_b_bf16(bcol[tc], k0, half);
#pragma unroll
    for (int tr = 0; tr < 2; ++tr)
#pragma unroll
      for (int tc = 0; tc < 4; ++tc) acc[tr][tc] = wmma_bf16(a[tr], b[tc], acc[tr][tc]);
  }

#pragma unroll
  for (int tr = 0; tr < 2; ++tr)
#pragma unroll
    for (int tc = 0; tc < 4; ++tc) {
      const int col = colBlk + tc * 16 + lidx;
      const float bv = bias[col];
#pragma unroll
      for (int r = 0; r < 8; ++r) {
        const int row = rowBlk + tr * 16 + r + 8 * half;
        const size_t idx = (size_t)row * kD + col;
        y[idx] = resid[idx] + acc[tr][tc][r] + bv;
      }
    }
}

// ---------------------------------------------------------------------------
// K6: LayerNorm over D=512 -> d_out (f32)
// ---------------------------------------------------------------------------
__global__ void __launch_bounds__(256)
ln_kernel(const float* __restrict__ y, const float* __restrict__ g,
          const float* __restrict__ b, float* __restrict__ out) {
  __shared__ float red[256];
  const size_t row = blockIdx.x;
  const float* p = y + row * kD;
  const int t = threadIdx.x;

  const float x0 = p[t];
  const float x1 = p[t + 256];

  red[t] = x0 + x1;
  __syncthreads();
  for (int s = 128; s > 0; s >>= 1) {
    if (t < s) red[t] += red[t + s];
    __syncthreads();
  }
  const float mean = red[0] * (1.0f / kD);
  __syncthreads();

  const float d0 = x0 - mean, d1 = x1 - mean;
  red[t] = d0 * d0 + d1 * d1;
  __syncthreads();
  for (int s = 128; s > 0; s >>= 1) {
    if (t < s) red[t] += red[t + s];
    __syncthreads();
  }
  const float var = red[0] * (1.0f / kD);
  const float rstd = rsqrtf(var + 1e-5f);

  out[row * kD + t] = d0 * rstd * g[t] + b[t];
  out[row * kD + t + 256] = d1 * rstd * g[t + 256] + b[t + 256];
}

// ---------------------------------------------------------------------------
// launch
// ---------------------------------------------------------------------------
extern "C" void kernel_launch(void* const* d_in, const int* in_sizes, int n_in,
                              void* d_out, int out_size, void* d_ws, size_t ws_size,
                              hipStream_t stream) {
  (void)in_sizes; (void)n_in; (void)out_size; (void)ws_size;

  const float* qxyz  = (const float*)d_in[0];
  const float* qfeat = (const float*)d_in[1];
  const float* kxyz  = (const float*)d_in[2];
  const float* kfeat = (const float*)d_in[3];
  const float* Wq = (const float*)d_in[4];
  const float* bq = (const float*)d_in[5];
  const float* Wk = (const float*)d_in[6];
  const float* bk = (const float*)d_in[7];
  const float* Wv = (const float*)d_in[8];
  const float* bv = (const float*)d_in[9];
  const float* Wo = (const float*)d_in[10];
  const float* bo = (const float*)d_in[11];
  const float* ln_g = (const float*)d_in[12];
  const float* ln_b = (const float*)d_in[13];

  char* ws = (char*)d_ws;
  const size_t szBND2 = (size_t)kB * kN * kD * sizeof(bf16_t);   // 16.8 MB
  const size_t szS    = (size_t)kB * kN * kM * sizeof(float);    // 134.2 MB
  const size_t szW2   = (size_t)kD * kD * sizeof(bf16_t);        // 0.5 MB

  bf16_t* Qb  = (bf16_t*)(ws);
  bf16_t* Kb  = (bf16_t*)(ws + szBND2);
  bf16_t* Vt  = (bf16_t*)(ws + 2 * szBND2);
  float*  S   = (float*)(ws + 3 * szBND2);       // bf16 P aliased in place
  bf16_t* att = (bf16_t*)(ws + 3 * szBND2 + szS);
  float*  y   = (float*)(ws + 4 * szBND2 + szS);
  bf16_t* Fq  = (bf16_t*)(ws + 4 * szBND2 + szS + (size_t)kRows * kD * 4);
  bf16_t* Fk  = (bf16_t*)((char*)Fq + szBND2);
  bf16_t* Wqb = (bf16_t*)((char*)Fk + szBND2);
  bf16_t* Wkb = (bf16_t*)((char*)Wqb + szW2);
  bf16_t* Wvb = (bf16_t*)((char*)Wkb + szW2);
  bf16_t* Wob = (bf16_t*)((char*)Wvb + szW2);

  const dim3 blk(256);
  const dim3 gGemm(kRows / 128, kD / 128);       // 128 x 4

  const int nFeat8 = kRows * kD / 8;             // 1,048,576
  const int nW8    = kD * kD / 8;                // 32,768
  cvt_kernel<<<dim3((nFeat8 + 255) / 256), blk, 0, stream>>>(qfeat, Fq, nFeat8);
  cvt_kernel<<<dim3((nFeat8 + 255) / 256), blk, 0, stream>>>(kfeat, Fk, nFeat8);
  cvt_kernel<<<dim3((nW8 + 255) / 256), blk, 0, stream>>>(Wq, Wqb, nW8);
  cvt_kernel<<<dim3((nW8 + 255) / 256), blk, 0, stream>>>(Wk, Wkb, nW8);
  cvt_kernel<<<dim3((nW8 + 255) / 256), blk, 0, stream>>>(Wv, Wvb, nW8);
  cvt_kernel<<<dim3((nW8 + 255) / 256), blk, 0, stream>>>(Wo, Wob, nW8);

  proj_kernel<false><<<gGemm, blk, 0, stream>>>(Fq, Wqb, bq, Qb, kN);
  proj_kernel<false><<<gGemm, blk, 0, stream>>>(Fk, Wkb, bk, Kb, kM);
  proj_kernel<true ><<<gGemm, blk, 0, stream>>>(Fk, Wvb, bv, Vt, kM);

  scores_kernel<<<dim3(kM / 256, kN / 64, kB), blk, 0, stream>>>(Qb, Kb, qxyz, kxyz, S);
  softmax_kernel<<<dim3(kRows), blk, 0, stream>>>(S);
  pv_kernel<<<gGemm, blk, 0, stream>>>((const bf16_t*)S, Vt, att);
  outproj_kernel<<<gGemm, blk, 0, stream>>>(att, Wob, bo, qfeat, y);
  ln_kernel<<<dim3(kRows), blk, 0, stream>>>(y, ln_g, ln_b, (float*)d_out);
}
